// Model_13391708029610
// MI455X (gfx1250) — compile-verified, matrix operand order checked by reading, not run on hardware
//
#include <hip/hip_runtime.h>
#include <cstdint>
#include <cstddef>

// ---------------- types for CDNA5 WMMA ----------------
typedef __bf16 bf16_t;
typedef bf16_t v16bf __attribute__((ext_vector_type(16)));
typedef float  v8f   __attribute__((ext_vector_type(8)));

union Frag {
    v16bf    v;
    uint32_t u[8];
};

static __device__ __forceinline__ uint16_t f2bf_bits(float x) {
    union { float f; uint32_t u; } t; t.f = x;
    uint32_t r = t.u + 0x7fffu + ((t.u >> 16) & 1u);   // round-to-nearest-even
    return (uint16_t)(r >> 16);
}
static __device__ __forceinline__ bf16_t f2bf(float x) {
    union { uint16_t s; bf16_t b; } o; o.s = f2bf_bits(x);
    return o.b;
}
static __device__ __forceinline__ uint32_t pack_bf2(float a, float b) {
    return (uint32_t)f2bf_bits(a) | ((uint32_t)f2bf_bits(b) << 16);
}

#define NNODES 50000
#define NEDGES 600000
#define W128   128

// gfx1250 async global->LDS copy path (ASYNCcnt-tracked), with safe fallback.
#if defined(__AMDGCN__) && __has_builtin(__builtin_amdgcn_global_load_async_to_lds_b32) && \
    __has_builtin(__builtin_amdgcn_s_wait_asynccnt)
#define USE_ASYNC_LDS 1
#define AS1 __attribute__((address_space(1)))
#define AS3 __attribute__((address_space(3)))
#else
#define USE_ASYNC_LDS 0
#endif

// ---------------- elementwise helpers ----------------
__global__ void k_zero_f(float* __restrict__ p, size_t n) {
    size_t i = (size_t)blockIdx.x * blockDim.x + threadIdx.x;
    if (i < n) p[i] = 0.0f;
}
__global__ void k_zero_i(int* __restrict__ p, size_t n) {
    size_t i = (size_t)blockIdx.x * blockDim.x + threadIdx.x;
    if (i < n) p[i] = 0;
}
__global__ void k_copy_i(const int* __restrict__ in, int* __restrict__ out, size_t n) {
    size_t i = (size_t)blockIdx.x * blockDim.x + threadIdx.x;
    if (i < n) out[i] = in[i];
}
__global__ void k_f32_to_bf16(const float* __restrict__ in, bf16_t* __restrict__ out, size_t n) {
    size_t i = (size_t)blockIdx.x * blockDim.x + threadIdx.x;
    if (i < n) out[i] = f2bf(in[i]);
}

// ---------------- CSR build (once per call; edge_index is constant) ------------------
__global__ void k_hist(const long long* __restrict__ ei, int* __restrict__ deg, int nE) {
    int e = blockIdx.x * blockDim.x + threadIdx.x;
    if (e >= nE) return;
    atomicAdd(&deg[(int)ei[(size_t)nE + e]], 1);
}

// single-workgroup exclusive scan: rowptr[0..n], rowptr[n] = total
__global__ void k_scan(const int* __restrict__ deg, int* __restrict__ rowptr, int n) {
    __shared__ int buf[1024];
    __shared__ int carry;
    if (threadIdx.x == 0) carry = 0;
    __syncthreads();
    for (int base = 0; base < n; base += 1024) {
        int i = base + threadIdx.x;
        int v = (i < n) ? deg[i] : 0;
        buf[threadIdx.x] = v;
        __syncthreads();
        for (int off = 1; off < 1024; off <<= 1) {
            int t = (threadIdx.x >= off) ? buf[threadIdx.x - off] : 0;
            __syncthreads();
            buf[threadIdx.x] += t;
            __syncthreads();
        }
        int incl = buf[threadIdx.x];
        if (i < n) rowptr[i] = carry + incl - v;   // exclusive
        __syncthreads();
        if (threadIdx.x == 1023) carry += buf[1023];
        __syncthreads();
    }
    if (threadIdx.x == 0) rowptr[n] = carry;
}

__global__ void k_fill(const long long* __restrict__ ei, int* __restrict__ cursor,
                       int* __restrict__ srclist, int nE) {
    int e = blockIdx.x * blockDim.x + threadIdx.x;
    if (e >= nE) return;
    int d = (int)ei[(size_t)nE + e];
    int s = (int)ei[e];
    int pos = atomicAdd(&cursor[d], 1);
    srclist[pos] = s;
}

// ---------------- neighbor mean via CSR gather (no float atomics) --------------------
// one wave per node: 32 lanes x float4 = 128 features; writes bf16 directly.
__global__ void k_gather_mean(const float* __restrict__ z, const int* __restrict__ rowptr,
                              const int* __restrict__ srclist, bf16_t* __restrict__ mb, int M) {
    int gw   = (int)(((size_t)blockIdx.x * blockDim.x + threadIdx.x) >> 5);
    int lane = threadIdx.x & 31;
    if (gw >= M) return;
    int beg = rowptr[gw], end = rowptr[gw + 1];
    float ax = 0.f, ay = 0.f, az = 0.f, aw = 0.f;
    for (int e = beg; e < end; ++e) {
        int s = srclist[e];
        float4 v = ((const float4*)(z + (size_t)s * W128))[lane];
        ax += v.x; ay += v.y; az += v.z; aw += v.w;
    }
    int   d   = end - beg;
    float inv = 1.0f / (float)(d > 1 ? d : 1);
    uint2* dst = (uint2*)(mb + (size_t)gw * W128);
    dst[lane] = make_uint2(pack_bf2(ax * inv, ay * inv), pack_bf2(az * inv, aw * inv));
}

// ---------------- encoder layer 1: h = relu(x @ W1^T + b1), K=7, N=256 ----------------
__global__ void k_enc1(const float* __restrict__ x, const float* __restrict__ W1,
                       const float* __restrict__ b1, float* __restrict__ h, int M) {
    int idx = blockIdx.x * blockDim.x + threadIdx.x;
    if (idx >= M * 256) return;
    int n = idx >> 8, o = idx & 255;
    const float* xr = x  + (size_t)n * 7;
    const float* wr = W1 + (size_t)o * 7;
    float acc = b1[o];
#pragma unroll
    for (int k = 0; k < 7; ++k) acc = fmaf(xr[k], wr[k], acc);
    h[idx] = fmaxf(acc, 0.0f);
}

// ---------------- WMMA GEMM: out = A@B^T (+A2@B2^T) + bias, optional relu ------------
// A/A2: [M,K] bf16 row-major; B/B2: [N,K] bf16 row-major; out: [M,N] f32.
// Block = 8 waves; one 16-row M-tile per block staged into LDS (shared by all 8 waves,
// async global->LDS when available); each wave owns one 16-col N-tile.
// M%16==0, K in {128,256}, (N/16)%8==0 at all call sites.
__global__ void k_gemm_wmma(const bf16_t* __restrict__ A,  const bf16_t* __restrict__ B,
                            const float*  __restrict__ bias,
                            const bf16_t* __restrict__ A2, const bf16_t* __restrict__ B2,
                            float* __restrict__ out, int M, int N, int K, int do_relu) {
    __shared__ uint32_t sA[2 * 16 * 129];      // max: two 16x256 bf16 tiles, padded rows

    const int lane  = threadIdx.x & 31;
    const int wave  = threadIdx.x >> 5;
    const int ntile = wave + (blockIdx.y << 3);
    const int mtile = blockIdx.x;

    const int kd     = K >> 1;                 // dwords per row
    const int ksh    = (K == 256) ? 7 : 6;     // log2(kd)
    const int stride = kd + 1;                 // padded LDS row stride (bank-conflict-free)
    uint32_t* s0 = sA;
    uint32_t* s1 = sA + 16 * stride;

    const uint32_t* Ad  = (const uint32_t*)A + (size_t)(mtile << 4) * kd;
    const uint32_t* A2d = A2 ? (const uint32_t*)A2 + (size_t)(mtile << 4) * kd : nullptr;

    // ---- stage A (and A2) tile into LDS, shared by all 8 waves ----
    for (int idx = threadIdx.x; idx < (kd << 4); idx += 256) {
        int r = idx >> ksh;
        int c = idx & (kd - 1);
#if USE_ASYNC_LDS
        __builtin_amdgcn_global_load_async_to_lds_b32(
            (AS1 int*)(Ad + idx), (AS3 int*)(s0 + r * stride + c), 0, 0);
        if (A2d)
            __builtin_amdgcn_global_load_async_to_lds_b32(
                (AS1 int*)(A2d + idx), (AS3 int*)(s1 + r * stride + c), 0, 0);
#else
        s0[r * stride + c] = Ad[idx];
        if (A2d) s1[r * stride + c] = A2d[idx];
#endif
    }
#if USE_ASYNC_LDS
    __builtin_amdgcn_s_wait_asynccnt(0);
#endif
    __syncthreads();

    const int r15 = lane & 15;
    const int kh2 = (lane >> 4) << 2;          // K-half dword offset (0 or 4), ISA fragment layout
    const int col = (ntile << 4) + r15;

    const uint32_t* ap = s0 + r15 * stride + kh2;
    const uint32_t* bp = (const uint32_t*)(B + (size_t)col * K) + kh2;

    v8f acc = {};
    for (int kb2 = 0; kb2 < kd; kb2 += 16) {   // 32 K-elements (16 dwords) per WMMA
        Frag a, b;
#pragma unroll
        for (int j = 0; j < 4; ++j) {
            a.u[j]     = ap[kb2 + j];          // K = kb + khalf + {0..7}
            a.u[j + 4] = ap[kb2 + 8 + j];      // K = kb + khalf + {16..23}
            b.u[j]     = bp[kb2 + j];
            b.u[j + 4] = bp[kb2 + 8 + j];
        }
        acc = __builtin_amdgcn_wmma_f32_16x16x32_bf16(false, a.v, false, b.v,
                                                      (short)0, acc, false, false);
    }

    if (A2 != nullptr) {   // fused second product (SAGE: lin_l(mean) + lin_r(z))
        const uint32_t* ap2 = s1 + r15 * stride + kh2;
        const uint32_t* bp2 = (const uint32_t*)(B2 + (size_t)col * K) + kh2;
        for (int kb2 = 0; kb2 < kd; kb2 += 16) {
            Frag a, b;
#pragma unroll
            for (int j = 0; j < 4; ++j) {
                a.u[j]     = ap2[kb2 + j];
                a.u[j + 4] = ap2[kb2 + 8 + j];
                b.u[j]     = bp2[kb2 + j];
                b.u[j + 4] = bp2[kb2 + 8 + j];
            }
            acc = __builtin_amdgcn_wmma_f32_16x16x32_bf16(false, a.v, false, b.v,
                                                          (short)0, acc, false, false);
        }
    }

    // C/D layout: lane holds col = ntile*16 + (lane&15); VGPR v holds row
    // M = mtile*16 + v + 8*(lane>>4).
    const float bv    = bias[col];
    const int   mbase = (mtile << 4) + ((lane >> 4) << 3);
#pragma unroll
    for (int v = 0; v < 8; ++v) {
        float r = acc[v] + bv;
        if (do_relu) r = fmaxf(r, 0.0f);
        out[(size_t)(mbase + v) * N + col] = r;
    }
}

// ---------------- batchnorm ----------------------------------------------------------
__global__ void k_bn_stats(const float* __restrict__ z, float* __restrict__ sums, int M) {
    int f  = threadIdx.x;                      // blockDim.x == 128
    int n0 = blockIdx.x * 128;
    int n1 = n0 + 128 < M ? n0 + 128 : M;
    float s = 0.0f, q = 0.0f;
    for (int n = n0; n < n1; ++n) {
        float v = z[(size_t)n * W128 + f];
        s += v; q += v * v;
    }
    atomicAdd(&sums[f], s);
    atomicAdd(&sums[W128 + f], q);
}

__global__ void k_bn_final(const float* __restrict__ sums, const float* __restrict__ g,
                           const float* __restrict__ b, float* __restrict__ ss, float invM) {
    int f = threadIdx.x;
    if (f >= W128) return;
    float m   = sums[f] * invM;
    float var = sums[W128 + f] * invM - m * m;
    float sc  = g[f] * rsqrtf(var + 1e-5f);
    ss[f]        = sc;
    ss[W128 + f] = b[f] - m * sc;
}

__global__ void k_bn_apply_relu(float* __restrict__ z, const float* __restrict__ ss, int M) {
    size_t i = (size_t)blockIdx.x * blockDim.x + threadIdx.x;
    if (i >= (size_t)M * W128) return;
    int f = (int)(i & (W128 - 1));
    z[i] = fmaxf(fmaf(z[i], ss[f], ss[W128 + f]), 0.0f);
}

// ---------------- decoder layer 2: out = h @ W2^T + b2, K=256, N=4 -------------------
__global__ void k_dec2(const float* __restrict__ h, const float* __restrict__ W2,
                       const float* __restrict__ b2, float* __restrict__ out, int M) {
    int idx = blockIdx.x * blockDim.x + threadIdx.x;
    if (idx >= M * 4) return;
    int n = idx >> 2, o = idx & 3;
    const float* hr = h  + (size_t)n * 256;
    const float* wr = W2 + (size_t)o * 256;
    float acc = b2[o];
#pragma unroll 8
    for (int k = 0; k < 256; ++k) acc = fmaf(hr[k], wr[k], acc);
    out[idx] = acc;
}

// =====================================================================================
extern "C" void kernel_launch(void* const* d_in, const int* in_sizes, int n_in,
                              void* d_out, int out_size, void* d_ws, size_t ws_size,
                              hipStream_t stream) {
    const float*     x      = (const float*)d_in[0];
    const long long* ei     = (const long long*)d_in[1];   // int64 [2, NEDGES]
    const float*     enc_W1 = (const float*)d_in[2];
    const float*     enc_b1 = (const float*)d_in[3];
    const float*     enc_W2 = (const float*)d_in[4];
    const float*     enc_b2 = (const float*)d_in[5];
    const float*     Wl     = (const float*)d_in[6];
    const float*     bl     = (const float*)d_in[7];
    const float*     Wr     = (const float*)d_in[8];
    const float*     gamma  = (const float*)d_in[9];
    const float*     beta   = (const float*)d_in[10];
    const float*     dec_W1 = (const float*)d_in[11];
    const float*     dec_b1 = (const float*)d_in[12];
    const float*     dec_W2 = (const float*)d_in[13];
    const float*     dec_b2 = (const float*)d_in[14];
    float* out = (float*)d_out;

    const int M = NNODES;

    // ---- workspace carve-up ----
    char* wp = (char*)d_ws;
    auto alloc = [&](size_t bytes) -> void* {
        void* p = (void*)wp;
        wp += (bytes + 255) & ~(size_t)255;
        return p;
    };
    float*  zA      = (float*)alloc((size_t)M * W128 * 4);
    float*  zB      = (float*)alloc((size_t)M * W128 * 4);
    float*  h32     = (float*)alloc((size_t)M * 256 * 4);
    bf16_t* zbf     = (bf16_t*)alloc((size_t)M * W128 * 2);
    bf16_t* meanbf  = (bf16_t*)alloc((size_t)M * W128 * 2);
    bf16_t* hbf     = (bf16_t*)alloc((size_t)M * 256 * 2);
    bf16_t* encW2bf = (bf16_t*)alloc((size_t)W128 * 256 * 2);
    bf16_t* Wlbf    = (bf16_t*)alloc((size_t)5 * W128 * W128 * 2);
    bf16_t* Wrbf    = (bf16_t*)alloc((size_t)5 * W128 * W128 * 2);
    bf16_t* decW1bf = (bf16_t*)alloc((size_t)256 * W128 * 2);
    int*    deg     = (int*)alloc((size_t)M * 4);
    int*    rowptr  = (int*)alloc((size_t)(M + 1) * 4);
    int*    cursor  = (int*)alloc((size_t)M * 4);
    int*    srclist = (int*)alloc((size_t)NEDGES * 4);
    float*  bnsums  = (float*)alloc(2 * W128 * 4);
    float*  bnss    = (float*)alloc(2 * W128 * 4);

    const int T = 256;
    auto g1 = [](size_t n, int t) { return (unsigned)((n + t - 1) / t); };

    // ---- convert weights to bf16 ----
    k_f32_to_bf16<<<g1((size_t)W128 * 256, T), T, 0, stream>>>(enc_W2, encW2bf, (size_t)W128 * 256);
    k_f32_to_bf16<<<g1((size_t)5 * W128 * W128, T), T, 0, stream>>>(Wl, Wlbf, (size_t)5 * W128 * W128);
    k_f32_to_bf16<<<g1((size_t)5 * W128 * W128, T), T, 0, stream>>>(Wr, Wrbf, (size_t)5 * W128 * W128);
    k_f32_to_bf16<<<g1((size_t)256 * W128, T), T, 0, stream>>>(dec_W1, decW1bf, (size_t)256 * W128);

    // ---- build CSR once (dst -> list of src); replaces 384M f32 atomics with ~1.2M int atomics
    k_zero_i<<<g1((size_t)M, T), T, 0, stream>>>(deg, (size_t)M);
    k_hist<<<g1((size_t)NEDGES, T), T, 0, stream>>>(ei, deg, NEDGES);
    k_scan<<<1, 1024, 0, stream>>>(deg, rowptr, M);
    k_copy_i<<<g1((size_t)M, T), T, 0, stream>>>(rowptr, cursor, (size_t)M);
    k_fill<<<g1((size_t)NEDGES, T), T, 0, stream>>>(ei, cursor, srclist, NEDGES);

    // ---- encoder ----
    k_enc1<<<g1((size_t)M * 256, T), T, 0, stream>>>(x, enc_W1, enc_b1, h32, M);
    k_f32_to_bf16<<<g1((size_t)M * 256, T), T, 0, stream>>>(h32, hbf, (size_t)M * 256);
    {   // z = hbf @ enc_W2^T + b2   (M x 128, K = 256)
        dim3 gd(M / 16, 1);
        k_gemm_wmma<<<gd, 256, 0, stream>>>(hbf, encW2bf, enc_b2, nullptr, nullptr,
                                            zA, M, W128, 256, 0);
    }

    float* zcur = zA;
    float* znxt = zB;

    // ---- 5 SAGE convs ----
    for (int i = 0; i < 5; ++i) {
        k_gather_mean<<<g1((size_t)M * 32, T), T, 0, stream>>>(zcur, rowptr, srclist, meanbf, M);
        k_f32_to_bf16<<<g1((size_t)M * W128, T), T, 0, stream>>>(zcur, zbf, (size_t)M * W128);
        {   // znxt = meanbf @ Wl[i]^T + bl[i] + zbf @ Wr[i]^T
            dim3 gd(M / 16, 1);
            k_gemm_wmma<<<gd, 256, 0, stream>>>(meanbf, Wlbf + (size_t)i * W128 * W128,
                                                bl + (size_t)i * W128,
                                                zbf, Wrbf + (size_t)i * W128 * W128,
                                                znxt, M, W128, W128, 0);
        }
        float* t = zcur; zcur = znxt; znxt = t;

        if (i < 4) {  // batchnorm (batch stats) + relu
            k_zero_f<<<1, 2 * W128, 0, stream>>>(bnsums, 2 * W128);
            k_bn_stats<<<g1((size_t)M, 128), 128, 0, stream>>>(zcur, bnsums, M);
            k_bn_final<<<1, 128, 0, stream>>>(bnsums, gamma + (size_t)i * W128,
                                              beta + (size_t)i * W128, bnss, 1.0f / (float)M);
            k_bn_apply_relu<<<g1((size_t)M * W128, T), T, 0, stream>>>(zcur, bnss, M);
        }
    }

    // ---- decoder ----
    k_f32_to_bf16<<<g1((size_t)M * W128, T), T, 0, stream>>>(zcur, zbf, (size_t)M * W128);
    {   // h32 = relu(zbf @ dec_W1^T + dec_b1)   (M x 256, K = 128)
        dim3 gd(M / 16, 2);
        k_gemm_wmma<<<gd, 256, 0, stream>>>(zbf, decW1bf, dec_b1, nullptr, nullptr,
                                            h32, M, 256, W128, 1);
    }
    k_dec2<<<g1((size_t)M * 4, T), T, 0, stream>>>(h32, dec_W2, dec_b2, out, M);
}
